// LeakyESNHead_5394478924031
// MI455X (gfx1250) — compile-verified, matrix operand order checked by reading, not run on hardware
//
#include <hip/hip_runtime.h>
#include <hip/hip_bf16.h>
#include <math.h>

#define D_IN     1024
#define N_STATE  2048
#define BATCH    8
#define TSTEPS   2048
#define KTOT     (N_STATE + D_IN)        // 3072 (h | x concat along K)
#define KHALF    (KTOT / 2)              // 1536 per K-partition
#define ROWS_PER_WG 32
#define NUM_WGS  (N_STATE / ROWS_PER_WG) // 64 persistent workgroups
#define THREADS  256

typedef __attribute__((ext_vector_type(16))) __bf16 v16bf;
typedef __attribute__((ext_vector_type(4)))  __bf16 v4bf;
typedef __attribute__((ext_vector_type(8)))  float  v8f;

__device__ __forceinline__ v4bf cvt4(float4 g) {
    v4bf r;
    r.x = (__bf16)g.x; r.y = (__bf16)g.y; r.z = (__bf16)g.z; r.w = (__bf16)g.w;
    return r;
}

__device__ __forceinline__ void ldfrag(v16bf& f, const __bf16* p) {
    ((uint4*)&f)[0] = *(const uint4*)(p);       // K = base + khalf*8 .. +8
    ((uint4*)&f)[1] = *(const uint4*)(p + 16);  // K = base + 16 + khalf*8 .. +8
}

__global__ void esn_init_ctr(unsigned int* ctr) {
    if (threadIdx.x == 0) *ctr = 0u;
}

__global__ __launch_bounds__(THREADS, 1)
void esn_persistent(const float* __restrict__ x,
                    const float* __restrict__ h0,
                    const float* __restrict__ Win,
                    const float* __restrict__ Wrec,
                    const float* __restrict__ a,
                    float* __restrict__ out,          // (B, T, N) -- also the h history
                    unsigned int* __restrict__ ctr)
{
    // LDS: weights [32][3072] bf16 (192 KB) + acts [16][3072] bf16 (96 KB) + 2 KB reduce
    extern __shared__ __align__(16) char smem_raw[];
    __bf16* wLds   = (__bf16*)smem_raw;
    __bf16* actLds = (__bf16*)(smem_raw + (size_t)ROWS_PER_WG * KTOT * 2);
    float*  redLds = (float*) (smem_raw + (size_t)(ROWS_PER_WG + 16) * KTOT * 2);

    const int tid  = threadIdx.x;
    const int wg   = blockIdx.x;
    const int n0   = wg * ROWS_PER_WG;
    const int lane = tid & 31;           // wave32
    const int wave = tid >> 5;

    // ---------------- one-time init: pin bf16 weight slice in LDS ----------------
    #pragma unroll 1
    for (int it = 0; it < (ROWS_PER_WG * N_STATE / 4) / THREADS; ++it) {   // 64
        int i  = (it * THREADS + tid) * 4;
        int nl = i >> 11;
        int k  = i & (N_STATE - 1);
        float4 g = *(const float4*)(Wrec + (size_t)(n0 + nl) * N_STATE + k);
        *(v4bf*)(wLds + (size_t)nl * KTOT + k) = cvt4(g);
    }
    #pragma unroll 1
    for (int it = 0; it < (ROWS_PER_WG * D_IN / 4) / THREADS; ++it) {      // 32
        int i  = (it * THREADS + tid) * 4;
        int nl = i >> 10;
        int k  = i & (D_IN - 1);
        float4 g = *(const float4*)(Win + (size_t)(n0 + nl) * D_IN + k);
        *(v4bf*)(wLds + (size_t)nl * KTOT + N_STATE + k) = cvt4(g);
    }
    // Zero padded activation rows 8..15 once (static zeros for the whole run)
    {
        v4bf z; z.x = (__bf16)0.f; z.y = (__bf16)0.f; z.z = (__bf16)0.f; z.w = (__bf16)0.f;
        #pragma unroll 1
        for (int it = 0; it < ((16 - BATCH) * KTOT / 4) / THREADS; ++it) { // 24
            int i = (it * THREADS + tid) * 4;
            *(v4bf*)(actLds + (size_t)BATCH * KTOT + i) = z;
        }
    }

    // Compute-wave roles: wave in [0,4): sub = n-subtile, kpart = K-half
    const int sub   = wave & 1;
    const int kpart = (wave >> 1) & 1;
    const int row   = lane & 15;
    const int khalf = lane >> 4;
    const int colN  = n0 + (sub << 4) + row;
    float a_n = 0.f;
    if (wave < 2) a_n = a[colN];
    const float one_m_a = 1.0f - a_n;

    const __bf16* aBase = actLds + (size_t)row * KTOT + kpart * KHALF + khalf * 8;
    const __bf16* bBase = wLds + (size_t)((sub << 4) + row) * KTOT + kpart * KHALF + khalf * 8;

    // ---------------------------- time-step scan ----------------------------
    #pragma unroll 1
    for (int t = 0; t < TSTEPS; ++t) {
        // Stage h(t-1) rows 0..7 as bf16 into actLds[b][0..2048)
        #pragma unroll 1
        for (int it = 0; it < (BATCH * N_STATE / 4) / THREADS; ++it) {     // 16
            int i = (it * THREADS + tid) * 4;
            int b = i >> 11;
            int n = i & (N_STATE - 1);
            float4 g;
            if (t == 0) g = *(const float4*)(h0 + (size_t)b * N_STATE + n);
            else        g = *(const float4*)(out + ((size_t)b * TSTEPS + (t - 1)) * N_STATE + n);
            *(v4bf*)(actLds + (size_t)b * KTOT + n) = cvt4(g);
        }
        // Stage x[:, t, :] as bf16 into actLds[b][2048..3072)
        #pragma unroll 1
        for (int it = 0; it < (BATCH * D_IN / 4) / THREADS; ++it) {        // 8
            int i = (it * THREADS + tid) * 4;
            int b = i >> 10;
            int d = i & (D_IN - 1);
            float4 g = *(const float4*)(x + ((size_t)b * TSTEPS + t) * D_IN + d);
            *(v4bf*)(actLds + (size_t)b * KTOT + N_STATE + d) = cvt4(g);
        }
        __syncthreads();

        // 4 compute waves: 2 n-subtiles x 2 K-halves, 48 WMMAs each,
        // double-buffered so DS loads for tile k+1 overlap WMMA of tile k.
        v8f c = {0.f, 0.f, 0.f, 0.f, 0.f, 0.f, 0.f, 0.f};
        if (wave < 4) {
            v16bf a0, b0, a1, b1;
            ldfrag(a0, aBase);
            ldfrag(b0, bBase);
            #pragma unroll 4
            for (int kc = 0; kc < KHALF - 64; kc += 64) {                  // 23 iters, 2 tiles each
                ldfrag(a1, aBase + kc + 32);
                ldfrag(b1, bBase + kc + 32);
                c = __builtin_amdgcn_wmma_f32_16x16x32_bf16(
                        false, a0, false, b0, (short)0, c, false, false);
                ldfrag(a0, aBase + kc + 64);
                ldfrag(b0, bBase + kc + 64);
                c = __builtin_amdgcn_wmma_f32_16x16x32_bf16(
                        false, a1, false, b1, (short)0, c, false, false);
            }
            // tail: tiles KHALF-64 and KHALF-32
            ldfrag(a1, aBase + KHALF - 32);
            ldfrag(b1, bBase + KHALF - 32);
            c = __builtin_amdgcn_wmma_f32_16x16x32_bf16(
                    false, a0, false, b0, (short)0, c, false, false);
            c = __builtin_amdgcn_wmma_f32_16x16x32_bf16(
                    false, a1, false, b1, (short)0, c, false, false);

            if (kpart == 1) {                  // publish partial sums for waves 0,1
                float* p = redLds + (size_t)((sub << 5) + lane) * 8;
                #pragma unroll
                for (int v = 0; v < 8; ++v) p[v] = c[v];
            }
        }
        __syncthreads();                       // uniform: partials visible

        if (wave < 2) {
            const float* p = redLds + (size_t)((sub << 5) + lane) * 8;
            #pragma unroll
            for (int v = 0; v < 8; ++v) c[v] += p[v];

            // C/D layout: lanes 0..15 hold M=0..7 in regs 0..7 -> the 8 real batches
            if (lane < 16) {
                const int n = colN;
                #pragma unroll
                for (int v = 0; v < 8; ++v) {
                    const int b = v;
                    float hp = (t == 0) ? h0[(size_t)b * N_STATE + n]
                                        : out[((size_t)b * TSTEPS + (t - 1)) * N_STATE + n];
                    float hn = one_m_a * hp + a_n * tanhf(c[v]);
                    out[((size_t)b * TSTEPS + t) * N_STATE + n] = hn;
                }
            }
        }

        // Device-wide barrier: release stores, signal, spin, rejoin
        __threadfence();
        __syncthreads();
        if (tid == 0) {
            __hip_atomic_fetch_add(ctr, 1u, __ATOMIC_RELEASE, __HIP_MEMORY_SCOPE_AGENT);
            const unsigned int target = (unsigned int)NUM_WGS * (unsigned int)(t + 1);
            while (__hip_atomic_load(ctr, __ATOMIC_ACQUIRE, __HIP_MEMORY_SCOPE_AGENT) < target) {
                __builtin_amdgcn_s_sleep(2);
            }
        }
        __syncthreads();
    }
}

extern "C" void kernel_launch(void* const* d_in, const int* in_sizes, int n_in,
                              void* d_out, int out_size, void* d_ws, size_t ws_size,
                              hipStream_t stream)
{
    const float* x    = (const float*)d_in[0];   // (B, T, D_IN)
    const float* h0   = (const float*)d_in[1];   // (B, N)
    const float* Win  = (const float*)d_in[2];   // (N, D_IN)
    const float* Wrec = (const float*)d_in[3];   // (N, N)
    const float* a    = (const float*)d_in[4];   // (N,)
    float* out        = (float*)d_out;           // (B, T, N)
    unsigned int* ctr = (unsigned int*)d_ws;

    esn_init_ctr<<<1, 64, 0, stream>>>(ctr);

    const size_t lds_bytes = (size_t)(ROWS_PER_WG + 16) * KTOT * 2 + 2 * 32 * 8 * sizeof(float);
    esn_persistent<<<NUM_WGS, THREADS, lds_bytes, stream>>>(x, h0, Win, Wrec, a, out, ctr);
}
